// HeadAttention_17935783428735
// MI455X (gfx1250) — matrix operand order repaired and checked by reading, not compiled, not run on hardware
//
#include <hip/hip_runtime.h>
#include <hip/hip_bf16.h>

typedef _Float16 f16;
typedef __attribute__((ext_vector_type(16))) _Float16 v16h;
typedef __attribute__((ext_vector_type(8)))  _Float16 f16x8;   // 16 B
typedef __attribute__((ext_vector_type(8)))  float    v8f;

#define EMB   1024
#define HEADD 64
#define SEQ   2048
#define BATCH 8
#define TOTROW (BATCH * SEQ)      // 16384
#define PSTRIDE 40                // f16 elems; 80B rows -> 16B-aligned, conflict-free

// Load 8 contiguous f16 (16 B) into fragment elements [o .. o+7].
__device__ __forceinline__ void ld8(v16h& d, int o, const f16* p) {
    const f16x8 t = *(const f16x8*)p;
#pragma unroll
    for (int e = 0; e < 8; ++e) d[o + e] = t[e];
}
__device__ __forceinline__ void ldfrag(v16h& d, const f16* p) {
    ld8(d, 0, p);
    ld8(d, 8, p + 8);
}

// ---------------------------------------------------------------------------
// Kernel 0: pre-swizzle weights f32 -> f16 into B-fragment strip layout.
// strip(p, k16, col)[e] = W_p[k16*16 + e][col]  (16 contiguous f16 per strip)
// ---------------------------------------------------------------------------
__global__ __launch_bounds__(256) void prep_weights_kernel(
    const float* __restrict__ Wq, const float* __restrict__ Wk,
    const float* __restrict__ Wv, f16* __restrict__ wsW)
{
    const int id = blockIdx.x * 256 + threadIdx.x;      // one strip per thread
    if (id >= 3 * 64 * 64) return;
    const int p   = id >> 12;          // /4096
    const int k16 = (id >> 6) & 63;
    const int col = id & 63;
    const float* W = (p == 0) ? Wq : (p == 1) ? Wk : Wv;
    f16* dst = wsW + (size_t)id * 16;
#pragma unroll
    for (int e = 0; e < 16; ++e)
        dst[e] = (f16)W[(k16 * 16 + e) * HEADD + col];
}

// ---------------------------------------------------------------------------
// Kernel 1: QKV projection, no LDS.  Each wave owns a 16-row tile of x and
// 12 WMMA accumulators (3 proj x 4 N-tiles).  B fragments double-buffered so
// the next 2x b128 loads overlap the current WMMA.
// q pre-scaled by 1/sqrt(H); V written transposed for kernel 2.
// ---------------------------------------------------------------------------
__global__ __launch_bounds__(256) void proj_qkv_kernel(
    const float* __restrict__ x, const f16* __restrict__ wsW,
    const float* __restrict__ bq, const float* __restrict__ bk,
    const float* __restrict__ bv,
    f16* __restrict__ qf, f16* __restrict__ kf, f16* __restrict__ vT)
{
    const int lane = threadIdx.x & 31;
    const int wave = threadIdx.x >> 5;
    const int row0 = blockIdx.x * 128 + wave * 16;
    const int m    = lane & 15;
    const int half = lane >> 4;
    const int n    = lane & 15;

    v8f acc[3][4];
#pragma unroll
    for (int p = 0; p < 3; ++p)
#pragma unroll
        for (int t = 0; t < 4; ++t) acc[p][t] = (v8f){};

    for (int kb = 0; kb < EMB; kb += 32) {
        // ---- A fragment: 16x32 tile of x, f32 -> f16 (4x float4 loads) ----
        v16h a;
        const float* xrow = x + (size_t)(row0 + m) * EMB + kb + 16 * half;
        const float4 x0 = *(const float4*)(xrow);
        const float4 x1 = *(const float4*)(xrow + 4);
        const float4 x2 = *(const float4*)(xrow + 8);
        const float4 x3 = *(const float4*)(xrow + 12);
        a[0]  = (f16)x0.x; a[1]  = (f16)x0.y; a[2]  = (f16)x0.z; a[3]  = (f16)x0.w;
        a[4]  = (f16)x1.x; a[5]  = (f16)x1.y; a[6]  = (f16)x1.z; a[7]  = (f16)x1.w;
        a[8]  = (f16)x2.x; a[9]  = (f16)x2.y; a[10] = (f16)x2.z; a[11] = (f16)x2.w;
        a[12] = (f16)x3.x; a[13] = (f16)x3.y; a[14] = (f16)x3.z; a[15] = (f16)x3.w;

        const int k16 = (kb >> 4) + half;       // 16-row strip index for this lane
        const f16* wstrip = wsW + (size_t)(k16 * 64 + n) * 16;  // + (p*64*64 + t*16)*16

        // ---- 12 WMMAs, double-buffered B fragments ----
        v16h fA, fB;
        ldfrag(fA, wstrip);                      // idx 0: p=0,t=0
#pragma unroll
        for (int idx = 0; idx < 12; ++idx) {
            if (idx + 1 < 12) {
                const int pn = (idx + 1) >> 2, tn = (idx + 1) & 3;
                const f16* wp = wstrip + (size_t)(pn * 64 * 64 + tn * 16) * 16;
                ldfrag((idx & 1) ? fA : fB, wp);
            }
            const int p = idx >> 2, t = idx & 3;
            acc[p][t] = __builtin_amdgcn_wmma_f32_16x16x32_f16(
                false, a, false, (idx & 1) ? fB : fA, (short)0, acc[p][t],
                false, false);
        }
    }

    // ---- epilogue: bias, q scale, stores (V transposed) ----
#pragma unroll
    for (int t = 0; t < 4; ++t) {
        const float bqv = bq[t * 16 + n];
        const float bkv = bk[t * 16 + n];
        const float bvv = bv[t * 16 + n];
#pragma unroll
        for (int r = 0; r < 8; ++r) {
            const int row = row0 + r + 8 * half;
            const size_t o = (size_t)row * HEADD + t * 16 + n;
            qf[o] = (f16)((acc[0][t][r] + bqv) * 0.125f);
            kf[o] = (f16)(acc[1][t][r] + bkv);
            vT[(size_t)(t * 16 + n) * TOTROW + row] = (f16)(acc[2][t][r] + bvv);
        }
    }
}

// ---------------------------------------------------------------------------
// Kernel 2: causal flash attention.  One wave per 16-query tile.
// Per 32-key block: load 4 K frags, 4 WMMA (S = Q K^T); issue all 4 V-frag
// loads BEFORE softmax so they fly under the shuffle/exp sequence; P goes
// f32 C-layout -> LDS f16 -> A-layout; 4 WMMA (O += P V).
// ---------------------------------------------------------------------------
__global__ __launch_bounds__(128) void flash_attn_kernel(
    const f16* __restrict__ qf, const f16* __restrict__ kf,
    const f16* __restrict__ vT, float* __restrict__ out)
{
    __shared__ f16 sP[4][16 * PSTRIDE];

    const int lane = threadIdx.x & 31;
    const int wave = threadIdx.x >> 5;
    const int b    = blockIdx.y;
    const int qm   = blockIdx.x * 64 + wave * 16;
    const int m    = lane & 15;
    const int half = lane >> 4;
    const int n    = lane & 15;

    const f16* qb = qf + (size_t)b * SEQ * HEADD;
    const f16* kb = kf + (size_t)b * SEQ * HEADD;
    const f16* vb = vT + (size_t)b * SEQ;          // column base within vT rows

    // Q fragments (16x64 = two 16x32 A fragments), resident all kernel
    v16h aq[2];
#pragma unroll
    for (int g = 0; g < 2; ++g) {
        const f16* qrow = qb + (size_t)(qm + m) * HEADD + g * 32 + 16 * half;
        ld8(aq[g], 0, qrow);
        ld8(aq[g], 8, qrow + 8);
    }

    v8f o0 = (v8f){}, o1 = (v8f){}, o2 = (v8f){}, o3 = (v8f){};
    float mstate[8], lstate[8];
#pragma unroll
    for (int r = 0; r < 8; ++r) { mstate[r] = -1e30f; lstate[r] = 0.0f; }

    f16* Pw = sP[wave];

    for (int j = 0; j < qm + 16; j += 32) {
        // ---- K fragments (one clause of 8x b128), then 4 score WMMAs ----
        v16h bk00, bk01, bk10, bk11;
        {
            const f16* kp0 = kb + (size_t)(j + n) * HEADD + 16 * half;
            const f16* kp1 = kb + (size_t)(j + 16 + n) * HEADD + 16 * half;
            ldfrag(bk00, kp0);        // g=0, keys j..j+15
            ldfrag(bk01, kp1);        // g=0, keys j+16..j+31
            ldfrag(bk10, kp0 + 32);   // g=1
            ldfrag(bk11, kp1 + 32);
        }
        v8f c0 = (v8f){}, c1 = (v8f){};
        c0 = __builtin_amdgcn_wmma_f32_16x16x32_f16(false, aq[0], false, bk00,
                                                    (short)0, c0, false, false);
        c1 = __builtin_amdgcn_wmma_f32_16x16x32_f16(false, aq[0], false, bk01,
                                                    (short)0, c1, false, false);
        c0 = __builtin_amdgcn_wmma_f32_16x16x32_f16(false, aq[1], false, bk10,
                                                    (short)0, c0, false, false);
        c1 = __builtin_amdgcn_wmma_f32_16x16x32_f16(false, aq[1], false, bk11,
                                                    (short)0, c1, false, false);

        // ---- V fragments issued NOW; consumed only after softmax ----
        v16h bv0, bv1, bv2, bv3;
        ldfrag(bv0, vb + (size_t)(0 * 16 + n) * TOTROW + j + 16 * half);
        ldfrag(bv1, vb + (size_t)(1 * 16 + n) * TOTROW + j + 16 * half);
        ldfrag(bv2, vb + (size_t)(2 * 16 + n) * TOTROW + j + 16 * half);
        ldfrag(bv3, vb + (size_t)(3 * 16 + n) * TOTROW + j + 16 * half);

        // ---- causal mask + online softmax (reduce across 16-lane halves) ----
#pragma unroll
        for (int r = 0; r < 8; ++r) {
            const int qrow = qm + r + 8 * half;
            c0[r] = (j + n      > qrow) ? -1e30f : c0[r];
            c1[r] = (j + 16 + n > qrow) ? -1e30f : c1[r];

            float mx = fmaxf(c0[r], c1[r]);
#pragma unroll
            for (int off = 8; off >= 1; off >>= 1)
                mx = fmaxf(mx, __shfl_xor(mx, off, 32));
            const float mnew  = fmaxf(mstate[r], mx);
            const float alpha = __expf(mstate[r] - mnew);
            mstate[r] = mnew;

            const float p0 = __expf(c0[r] - mnew);
            const float p1 = __expf(c1[r] - mnew);
            c0[r] = p0; c1[r] = p1;

            float rs = p0 + p1;
#pragma unroll
            for (int off = 8; off >= 1; off >>= 1)
                rs += __shfl_xor(rs, off, 32);
            lstate[r] = lstate[r] * alpha + rs;

            o0[r] *= alpha; o1[r] *= alpha; o2[r] *= alpha; o3[r] *= alpha;
        }

        // ---- P: f32 C-layout -> LDS f16 -> A-layout fragment (2x b128) ----
#pragma unroll
        for (int r = 0; r < 8; ++r) {
            const int row = r + 8 * half;
            Pw[row * PSTRIDE + n]      = (f16)c0[r];
            Pw[row * PSTRIDE + 16 + n] = (f16)c1[r];
        }
        asm volatile("s_wait_dscnt 0" ::: "memory");

        v16h ap;
        {
            const f16* pp = Pw + m * PSTRIDE + 16 * half;
            ld8(ap, 0, pp);
            ld8(ap, 8, pp + 8);
        }

        // ---- O += P @ V ----
        o0 = __builtin_amdgcn_wmma_f32_16x16x32_f16(false, ap, false, bv0,
                                                    (short)0, o0, false, false);
        o1 = __builtin_amdgcn_wmma_f32_16x16x32_f16(false, ap, false, bv1,
                                                    (short)0, o1, false, false);
        o2 = __builtin_amdgcn_wmma_f32_16x16x32_f16(false, ap, false, bv2,
                                                    (short)0, o2, false, false);
        o3 = __builtin_amdgcn_wmma_f32_16x16x32_f16(false, ap, false, bv3,
                                                    (short)0, o3, false, false);
    }

    // ---- normalize and store fp32 output ----
    float* ob = out + (size_t)b * SEQ * HEADD;
#pragma unroll
    for (int r = 0; r < 8; ++r) {
        const float inv = 1.0f / lstate[r];
        const int row = qm + r + 8 * half;
        float* orow = ob + (size_t)row * HEADD + n;
        orow[0]  = o0[r] * inv;
        orow[16] = o1[r] * inv;
        orow[32] = o2[r] * inv;
        orow[48] = o3[r] * inv;
    }
}

// ---------------------------------------------------------------------------
extern "C" void kernel_launch(void* const* d_in, const int* in_sizes, int n_in,
                              void* d_out, int out_size, void* d_ws, size_t ws_size,
                              hipStream_t stream) {
    const float* x  = (const float*)d_in[0];
    const float* Wk = (const float*)d_in[1];
    const float* bk = (const float*)d_in[2];
    const float* Wq = (const float*)d_in[3];
    const float* bq = (const float*)d_in[4];
    const float* Wv = (const float*)d_in[5];
    const float* bv = (const float*)d_in[6];

    const size_t per = (size_t)TOTROW * HEADD;   // 1,048,576 elems (2 MB f16)
    f16* qf  = (f16*)d_ws;
    f16* kf  = qf + per;
    f16* vT  = kf + per;
    f16* wsW = vT + per;                         // 196,608 elems (384 KB)

    prep_weights_kernel<<<dim3(48), dim3(256), 0, stream>>>(Wq, Wk, Wv, wsW);

    proj_qkv_kernel<<<dim3(TOTROW / 128), dim3(256), 0, stream>>>(
        x, wsW, bq, bk, bv, qf, kf, vT);

    flash_attn_kernel<<<dim3(SEQ / 64, BATCH), dim3(128), 0, stream>>>(
        qf, kf, vT, (float*)d_out);
}
